// SimpleRNN_41111426957822
// MI455X (gfx1250) — matrix operand order repaired and checked by reading, not compiled, not run on hardware
//
#include <hip/hip_runtime.h>
#include <math.h>

#define H 256
#define SW 264                     // padded LDS row stride in f16 elems (528B = 132 dwords ≡ 4 mod 64 banks)
#define ROWS_PER_WAVE 32
#define WAVES_PER_WG 8
#define ROWS_PER_WG (ROWS_PER_WAVE * WAVES_PER_WG)   // 256
#define NBLK (H / 16)              // 16
#define KBLK (H / 32)              // 8
#define BATCH 65536

typedef _Float16 v8h  __attribute__((ext_vector_type(8)));
typedef _Float16 v16h __attribute__((ext_vector_type(16)));
typedef float    v8f  __attribute__((ext_vector_type(8)));

// A fragment: 16x32 f16. Lane l (<16) holds row l, K = [kb*32+0..7] and [kb*32+16..23];
// lane l (>=16) holds row l-16, K = [kb*32+8..15] and [kb*32+24..31]. Two ds_load_b128.
__device__ __forceinline__ v16h load_a_frag(const _Float16* hrow, int lane, int kb) {
    const int r    = lane & 15;
    const int koff = kb * 32 + ((lane < 16) ? 0 : 8);
    const _Float16* p = hrow + r * SW + koff;
    v8h lo = *(const v8h*)(p);
    v8h hi = *(const v8h*)(p + 16);
    v16h a;
#pragma unroll
    for (int i = 0; i < 8; ++i) { a[i] = lo[i]; a[i + 8] = hi[i]; }
    return a;
}

// B fragment: 32x16 f16 (KxN). Lane l (<16) holds column n=l, K = kb*32+[0..7] and [16..23];
// lane l (>=16) holds column n=l-16, K = kb*32+[8..15] and [24..31]. wt is N-major: wt[n*SW + k].
__device__ __forceinline__ v16h load_b_frag(const _Float16* wt, int lane, int kb, int nb) {
    const int n     = lane & 15;
    const int kbase = kb * 32 + ((lane < 16) ? 0 : 8);
    const _Float16* p = wt + (nb * 16 + n) * SW + kbase;
    v8h lo = *(const v8h*)(p);
    v8h hi = *(const v8h*)(p + 16);
    v16h b;
#pragma unroll
    for (int i = 0; i < 8; ++i) { b[i] = lo[i]; b[i + 8] = hi[i]; }
    return b;
}

__device__ __forceinline__ v8f wmma_f16(v16h a, v16h b, v8f c) {
    return __builtin_amdgcn_wmma_f32_16x16x32_f16(false, a, false, b, (short)0, c, false, false);
}

// Hardware TRANS tanh: co-executes with XDL WMMA (1 op vs ~15 for ocml tanhf).
__device__ __forceinline__ float fast_tanh(float x) {
    float r;
    asm("v_tanh_f32 %0, %1" : "=v"(r) : "v"(x));
    return r;
}

// Software-pipelined wave GEMM: 32x256 tile (A in regs) times 256x256 (B in LDS).
// B fragments double-buffered in registers so ds_load overlaps v_wmma.
template <typename Epilogue>
__device__ __forceinline__ void wave_gemm(const _Float16* wt, const v16h* A0, const v16h* A1,
                                          int lane, Epilogue epi) {
    v16h bcur = load_b_frag(wt, lane, 0, 0);
    for (int nb = 0; nb < NBLK; ++nb) {
        v8f c0 = {}, c1 = {};
#pragma unroll
        for (int kb = 0; kb < KBLK; ++kb) {
            const int nkb = (kb + 1) & (KBLK - 1);
            const int nnb = (kb == KBLK - 1) ? ((nb + 1) & (NBLK - 1)) : nb;
            v16h bnext = load_b_frag(wt, lane, nkb, nnb);   // prefetch while WMMA runs
            c0 = wmma_f16(A0[kb], bcur, c0);
            c1 = wmma_f16(A1[kb], bcur, c1);
            bcur = bnext;
        }
        epi(nb, c0, c1);
    }
}

extern "C" __global__ __launch_bounds__(256, 1)
void rnn_persistent_kernel(const float* __restrict__ x,
                           const float* __restrict__ W_hh,
                           const float* __restrict__ fc1_w,
                           const float* __restrict__ fc1_b,
                           const float* __restrict__ fc2_w,
                           const float* __restrict__ fc2_b,
                           const int*   __restrict__ steps_p,
                           float* __restrict__ out,     // [B*H]
                           float* __restrict__ h_out)   // [B*H]
{
    extern __shared__ _Float16 smem[];
    _Float16* wt = smem;                                       // H * SW  (132 KB)
    const int tid  = threadIdx.x;
    const int wave = tid >> 5;
    const int lane = tid & 31;
    _Float16* hbuf = smem + H * SW + wave * (ROWS_PER_WAVE * SW);  // wave-private 32xSW
    const long rowg = (long)blockIdx.x * ROWS_PER_WG + (long)wave * ROWS_PER_WAVE;

    const int nlane = lane & 15;
    const int mb    = (lane < 16) ? 0 : 8;

    // ---- Phase 1: Wt = fc1_w (already N-major: B(k,n) = fc1_w[n*H + k]) ----
    for (int i = tid; i < H * H; i += blockDim.x) {
        int n = i >> 8, k = i & 255;
        wt[n * SW + k] = (_Float16)fc1_w[i];
    }
    // load x tile (f32 -> f16) into wave-private region
    for (int i = lane; i < ROWS_PER_WAVE * H; i += 32) {
        int r = i >> 8, c = i & 255;
        hbuf[r * SW + c] = (_Float16)x[(rowg + r) * H + c];
    }
    __syncthreads();

    v16h A0[KBLK], A1[KBLK];

    // h0 = x @ fc1_w^T + fc1_b   (no activation)
#pragma unroll
    for (int kb = 0; kb < KBLK; ++kb) {
        A0[kb] = load_a_frag(hbuf, lane, kb);
        A1[kb] = load_a_frag(hbuf + 16 * SW, lane, kb);
    }
    wave_gemm(wt, A0, A1, lane, [&](int nb, v8f c0, v8f c1) {
        const float bias = fc1_b[nb * 16 + nlane];
        const int   n    = nb * 16 + nlane;
#pragma unroll
        for (int j = 0; j < 8; ++j) {
            hbuf[(mb + j) * SW + n]      = (_Float16)(c0[j] + bias);
            hbuf[(16 + mb + j) * SW + n] = (_Float16)(c1[j] + bias);
        }
    });
    __syncthreads();

    // ---- Phase 2: Wt = W_hh^T  (B(k,n) = W_hh[k*H + n] -> wt[n*SW + k]) ----
    for (int i = tid; i < H * H; i += blockDim.x) {
        int k = i >> 8, n = i & 255;
        wt[n * SW + k] = (_Float16)W_hh[i];
    }
    __syncthreads();

    const int steps = *steps_p;
    for (int s = 0; s < steps; ++s) {
        // snapshot full A tile into registers -> in-place LDS update is safe
#pragma unroll
        for (int kb = 0; kb < KBLK; ++kb) {
            A0[kb] = load_a_frag(hbuf, lane, kb);
            A1[kb] = load_a_frag(hbuf + 16 * SW, lane, kb);
        }
        wave_gemm(wt, A0, A1, lane, [&](int nb, v8f c0, v8f c1) {
            const int n = nb * 16 + nlane;
#pragma unroll
            for (int j = 0; j < 8; ++j) {
                hbuf[(mb + j) * SW + n]      = (_Float16)fast_tanh(c0[j]);
                hbuf[(16 + mb + j) * SW + n] = (_Float16)fast_tanh(c1[j]);
            }
        });
    }

    // ---- write final h (f32) ----
    for (int i = lane; i < ROWS_PER_WAVE * H; i += 32) {
        int r = i >> 8, c = i & 255;
        h_out[(rowg + r) * H + c] = (float)hbuf[r * SW + c];
    }
    __syncthreads();

    // ---- Phase 3: Wt = fc2_w (N-major already) ----
    for (int i = tid; i < H * H; i += blockDim.x) {
        int n = i >> 8, k = i & 255;
        wt[n * SW + k] = (_Float16)fc2_w[i];
    }
    __syncthreads();

    // out = h @ fc2_w^T + fc2_b, streamed to global
#pragma unroll
    for (int kb = 0; kb < KBLK; ++kb) {
        A0[kb] = load_a_frag(hbuf, lane, kb);
        A1[kb] = load_a_frag(hbuf + 16 * SW, lane, kb);
    }
    wave_gemm(wt, A0, A1, lane, [&](int nb, v8f c0, v8f c1) {
        const float bias = fc2_b[nb * 16 + nlane];
        const int   n    = nb * 16 + nlane;
#pragma unroll
        for (int j = 0; j < 8; ++j) {
            out[(rowg + mb + j) * H + n]      = c0[j] + bias;
            out[(rowg + 16 + mb + j) * H + n] = c1[j] + bias;
        }
    });
}

extern "C" void kernel_launch(void* const* d_in, const int* in_sizes, int n_in,
                              void* d_out, int out_size, void* d_ws, size_t ws_size,
                              hipStream_t stream) {
    (void)in_sizes; (void)n_in; (void)out_size; (void)d_ws; (void)ws_size;
    const float* x     = (const float*)d_in[0];
    const float* W_hh  = (const float*)d_in[1];
    const float* fc1_w = (const float*)d_in[2];
    const float* fc1_b = (const float*)d_in[3];
    const float* fc2_w = (const float*)d_in[4];
    const float* fc2_b = (const float*)d_in[5];
    const int*   steps = (const int*)d_in[6];

    float* out   = (float*)d_out;
    float* h_out = out + (size_t)BATCH * H;

    const size_t shmem = (size_t)(H * SW + ROWS_PER_WG * SW) * sizeof(_Float16); // 264 KB
    hipFuncSetAttribute((const void*)rnn_persistent_kernel,
                        hipFuncAttributeMaxDynamicSharedMemorySize, (int)shmem);

    dim3 grid(BATCH / ROWS_PER_WG);   // 256 workgroups
    dim3 block(32 * WAVES_PER_WG);    // 256 threads = 8 waves
    rnn_persistent_kernel<<<grid, block, shmem, stream>>>(
        x, W_hh, fc1_w, fc1_b, fc2_w, fc2_b, steps, out, h_out);
}
